// SA_WSLFA_5583457485366
// MI455X (gfx1250) — matrix operand order repaired and checked by reading, not compiled, not run on hardware
//
#include <hip/hip_runtime.h>
#include <math.h>

// ---------------- problem constants ----------------
#define BB     2
#define NN     8192
#define MM     4096
#define KNbr   32
#define CIN    64
#define COUT   128
#define DCAT   67           // 3 + CIN
#define DALPHA 195          // DCAT + COUT
#define DCATP  96           // padded to 3 x 32
#define DAP    224          // padded to 7 x 32
#define PP     (BB*MM*KNbr) // 262144 points  (== 1<<18)
#define BMTOT  (BB*MM)      // 8192           (== 1<<13)
#define EPSBN  1e-5f

typedef __attribute__((ext_vector_type(16))) __bf16        v16bf;
typedef __attribute__((ext_vector_type(8)))  float         v8f;
typedef __attribute__((ext_vector_type(4)))  unsigned int  v4u;
typedef __attribute__((ext_vector_type(8)))  unsigned int  v8u;

__device__ __forceinline__ unsigned short f2bf(float x) {
    unsigned u = __builtin_bit_cast(unsigned, x);
    unsigned r = u + 0x7FFFu + ((u >> 16) & 1u);   // round-to-nearest-even
    return (unsigned short)(r >> 16);
}

// ---------------- 1) centers ----------------
__global__ void wslfa_centers(const float* __restrict__ xyz,
                              float* __restrict__ centers,
                              float* __restrict__ out) {
    int t = blockIdx.x * blockDim.x + threadIdx.x;      // over B*M
    if (t >= BMTOT) return;
    int b = t >> 12, m = t & (MM - 1);
    const float step = (float)(NN - 1) / (float)(MM - 1);
    int idx = (int)((float)m * step);                   // linspace().astype(int32)
    const float* src = xyz + ((size_t)b * NN + idx) * 3;
    float cx = src[0], cy = src[1], cz = src[2];
    centers[t*3+0] = cx; centers[t*3+1] = cy; centers[t*3+2] = cz;
    out[t*3+0] = cx; out[t*3+1] = cy; out[t*3+2] = cz;  // output #1
}

// ---------------- 2) KNN: one wave32 per center ----------------
__global__ __launch_bounds__(32) void wslfa_knn(const float* __restrict__ xyz,
                                                const float* __restrict__ centers,
                                                int* __restrict__ idx_knn) {
    __shared__ float sd[NN];                            // 32 KB of the 320 KB WGP LDS
    int bm = blockIdx.x, b = bm >> 12, lane = threadIdx.x;
    float cx = centers[bm*3+0], cy = centers[bm*3+1], cz = centers[bm*3+2];
    float c2 = cx*cx + cy*cy + cz*cz;
    const float* xb = xyz + (size_t)b * NN * 3;
    for (int n = lane; n < NN; n += 32) {
        float x = xb[n*3+0], y = xb[n*3+1], z = xb[n*3+2];
        float x2 = x*x + y*y + z*z;
        sd[n] = c2 + x2 - 2.0f * (cx*x + cy*y + cz*z); // same formula as reference
    }
    __syncthreads();
    int myIdx = 0;
    for (int k = 0; k < KNbr; ++k) {
        float bv = __builtin_inff(); int bi = NN;
        for (int n = lane; n < NN; n += 32) {          // bank-conflict-free stride
            float v = sd[n];
            if (v < bv) { bv = v; bi = n; }            // strict < keeps lowest index
        }
        for (int off = 16; off > 0; off >>= 1) {       // wave argmin, stable tie-break
            float ov = __shfl_xor(bv, off, 32);
            int   oi = __shfl_xor(bi, off, 32);
            if (ov < bv || (ov == bv && oi < bi)) { bv = ov; bi = oi; }
        }
        if (lane == k) myIdx = bi;
        if (lane == 0) sd[bi] = __builtin_inff();      // exclude from later rounds
        __syncthreads();
    }
    idx_knn[bm * KNbr + lane] = myIdx;
}

// ---------------- 3) build bf16 activation rows [point][DAP] ----------------
// c in [0,3): local_xyz   c in [3,67): gathered feat   c in [67,224): zero (rewritten later)
__global__ void wslfa_cat(const float* __restrict__ xyz,
                          const float* __restrict__ feat,
                          const float* __restrict__ centers,
                          const int*   __restrict__ idx_knn,
                          unsigned short* __restrict__ act) {
    int p = blockIdx.x * blockDim.x + threadIdx.x;
    if (p >= PP) return;
    int bm = p >> 5, b = bm >> 12;
    int idx = idx_knn[p];
    unsigned short* row = act + (size_t)p * DAP;
    const float* c = centers + (size_t)bm * 3;
    const float* x = xyz + ((size_t)b * NN + idx) * 3;
    row[0] = f2bf(x[0] - c[0]);
    row[1] = f2bf(x[1] - c[1]);
    row[2] = f2bf(x[2] - c[2]);
    const float* f = feat + (size_t)b * CIN * NN + idx;
    #pragma unroll 4
    for (int cc = 0; cc < CIN; ++cc) row[3 + cc] = f2bf(f[(size_t)cc * NN]);
    for (int cc = DCAT; cc < DAP; ++cc) row[cc] = 0;   // re-zero pads every call
}

// ---------------- 4) weights -> padded bf16 ----------------
__global__ void wslfa_wprep(const float* __restrict__ Wf, const float* __restrict__ Wa,
                            unsigned short* __restrict__ Wfb, unsigned short* __restrict__ Wab) {
    int t = blockIdx.x * blockDim.x + threadIdx.x;      // COUT * DAP
    if (t >= COUT * DAP) return;
    int o = t / DAP, c = t % DAP;
    Wab[o * DAP + c] = (c < DALPHA) ? f2bf(Wa[o * DALPHA + c]) : (unsigned short)0;
    if (c < DCATP) Wfb[o * DCATP + c] = (c < DCAT) ? f2bf(Wf[o * DCAT + c]) : (unsigned short)0;
}

// ---------------- 5) WMMA GEMM: Y[o][p] = sum_c W[o][c]*Act[p][c] + bias[o] ----------------
// Each wave32 -> one 16(chan) x 16(point) tile via v_wmma_f32_16x16x32_bf16.
// A (16x32 bf16): lane m = lane&15, K-half = lane>>4; VGPR0..3 = K 8h..8h+7, VGPR4..7 = K 16+8h..+7
// B (32x16 bf16): lane n = lane&15, K-half = lane>>4; VGPR0..3 = K 8h..8h+7, VGPR4..7 = K 16+8h..+7 of column n
__global__ __launch_bounds__(256) void wslfa_gemm(const unsigned short* __restrict__ act, int bstride,
                                                  const unsigned short* __restrict__ W, int wstride,
                                                  int nChunks,
                                                  const float* __restrict__ bias,
                                                  float* __restrict__ Y) {
    int wave = threadIdx.x >> 5, lane = threadIdx.x & 31;
    int n = lane & 15, kh = lane >> 4;
    int chanBlk = wave;                                  // 8 waves -> 128 channels
    size_t ptBase = (size_t)blockIdx.x * 16;
    const unsigned short* brow = act + (ptBase + n) * (size_t)bstride;
    const unsigned short* wrow = W + (size_t)(chanBlk * 16 + n) * wstride;

    v8f acc = {0.f,0.f,0.f,0.f,0.f,0.f,0.f,0.f};
    for (int ch = 0; ch < nChunks; ++ch) {
        int cb = ch * 32;
        v4u alo = *(const v4u*)(wrow + cb + 8 * kh);
        v4u ahi = *(const v4u*)(wrow + cb + 16 + 8 * kh);
        v8u au  = __builtin_shufflevector(alo, ahi, 0,1,2,3,4,5,6,7);
        v4u blo = *(const v4u*)(brow + cb + 8 * kh);
        v4u bhi = *(const v4u*)(brow + cb + 16 + 8 * kh);
        v8u bu  = __builtin_shufflevector(blo, bhi, 0,1,2,3,4,5,6,7);
        v16bf a = __builtin_bit_cast(v16bf, au);
        v16bf bmat = __builtin_bit_cast(v16bf, bu);
        acc = __builtin_amdgcn_wmma_f32_16x16x32_bf16(false, a, false, bmat,
                                                      (short)0, acc, false, false);
    }
    // C/D layout: VGPR r -> chan r + 8*kh, column n
    #pragma unroll
    for (int r = 0; r < 8; ++r) {
        int chan = chanBlk * 16 + 8 * kh + r;
        Y[(size_t)chan * PP + ptBase + n] = acc[r] + bias[chan];
    }
}

// ---------------- 6) BN stats (training-mode, population var) ----------------
__global__ __launch_bounds__(256) void wslfa_bnstats(const float* __restrict__ Y,
                                                     float* __restrict__ mu,
                                                     float* __restrict__ invstd) {
    __shared__ float ss[256], ss2[256];
    int chan = blockIdx.x, tid = threadIdx.x;
    const float* y = Y + (size_t)chan * PP;
    float s = 0.f, s2 = 0.f;
    for (size_t i = tid; i < PP; i += 256) { float v = y[i]; s += v; s2 += v * v; }
    ss[tid] = s; ss2[tid] = s2; __syncthreads();
    for (int off = 128; off > 0; off >>= 1) {
        if (tid < off) { ss[tid] += ss[tid+off]; ss2[tid] += ss2[tid+off]; }
        __syncthreads();
    }
    if (tid == 0) {
        float m = ss[0] / (float)PP;
        float v = ss2[0] / (float)PP - m * m;
        mu[chan] = m;
        invstd[chan] = rsqrtf(v + EPSBN);
    }
}

// ---------------- 7) BN apply + ReLU in place (f_prime) ----------------
__global__ __launch_bounds__(256) void wslfa_bnapply(float* __restrict__ Y,
                                                     const float* __restrict__ mu,
                                                     const float* __restrict__ invstd,
                                                     const float* __restrict__ g,
                                                     const float* __restrict__ be) {
    size_t i = (size_t)blockIdx.x * 256 + threadIdx.x;  // COUT*PP elements
    int chan = (int)(i >> 18);                          // PP == 1<<18
    float v = Y[i];
    v = (v - mu[chan]) * invstd[chan] * g[chan] + be[chan];
    Y[i] = fmaxf(v, 0.f);
}

// ---------------- 8) f' - mean_K(f') -> bf16 into act[:, 67..194] ----------------
__global__ __launch_bounds__(256) void wslfa_center(const float* __restrict__ F,
                                                    unsigned short* __restrict__ act) {
    int t = blockIdx.x * 256 + threadIdx.x;             // COUT * BMTOT
    if (t >= COUT * BMTOT) return;
    int o = t >> 13, bm = t & (BMTOT - 1);
    const float* f = F + (size_t)o * PP + (size_t)bm * KNbr;
    float vals[KNbr]; float s = 0.f;
    #pragma unroll
    for (int k = 0; k < KNbr; ++k) { vals[k] = f[k]; s += vals[k]; }
    float mn = s * (1.0f / (float)KNbr);
    unsigned short* col = act + (size_t)bm * KNbr * DAP + DCAT + o;
    #pragma unroll
    for (int k = 0; k < KNbr; ++k) col[(size_t)k * DAP] = f2bf(vals[k] - mn);
}

// ---------------- 9) alpha BN+ReLU, softmax over K, weighted sum (one wave per (bm,o)) ----
__global__ __launch_bounds__(256) void wslfa_final(const float* __restrict__ Y2,
                                                   const float* __restrict__ F,
                                                   const float* __restrict__ mu,
                                                   const float* __restrict__ is,
                                                   const float* __restrict__ g,
                                                   const float* __restrict__ be,
                                                   float* __restrict__ out) {
    int wave = threadIdx.x >> 5, lane = threadIdx.x & 31;
    int t = blockIdx.x * 8 + wave;                      // COUT*BMTOT waves
    int o = t >> 13, bm = t & (BMTOT - 1);
    size_t p = (size_t)bm * KNbr + lane;
    float a = (Y2[(size_t)o * PP + p] - mu[o]) * is[o] * g[o] + be[o];
    a = fmaxf(a, 0.f);
    float mx = a;
    for (int off = 16; off > 0; off >>= 1) mx = fmaxf(mx, __shfl_xor(mx, off, 32));
    float e = expf(a - mx);
    float s = e;
    for (int off = 16; off > 0; off >>= 1) s += __shfl_xor(s, off, 32);
    float r = (e / s) * F[(size_t)o * PP + p];
    for (int off = 16; off > 0; off >>= 1) r += __shfl_xor(r, off, 32);
    if (lane == 0) {
        int b = bm >> 12, m = bm & (MM - 1);
        out[(size_t)BMTOT * 3 + ((size_t)(b * COUT + o)) * MM + m] = r;
    }
}

extern "C" void kernel_launch(void* const* d_in, const int* in_sizes, int n_in,
                              void* d_out, int out_size, void* d_ws, size_t ws_size,
                              hipStream_t stream) {
    const float* xyz  = (const float*)d_in[0];
    const float* feat = (const float*)d_in[1];
    const float* Wf   = (const float*)d_in[2];
    const float* bf_  = (const float*)d_in[3];
    const float* gf   = (const float*)d_in[4];
    const float* bef  = (const float*)d_in[5];
    const float* Wa   = (const float*)d_in[6];
    const float* ba_  = (const float*)d_in[7];
    const float* ga   = (const float*)d_in[8];
    const float* bea  = (const float*)d_in[9];
    float* out = (float*)d_out;

    char* ws = (char*)d_ws;
    size_t off = 0;
    auto carve = [&](size_t bytes) -> void* {
        void* p = ws + off;
        off = (off + bytes + 255) & ~(size_t)255;
        return p;
    };
    float*          centers = (float*)carve((size_t)BMTOT * 3 * 4);
    int*            idxknn  = (int*)carve((size_t)BMTOT * KNbr * 4);
    float*          mu1     = (float*)carve(COUT * 4);
    float*          is1     = (float*)carve(COUT * 4);
    float*          mu2     = (float*)carve(COUT * 4);
    float*          is2     = (float*)carve(COUT * 4);
    unsigned short* Wfb     = (unsigned short*)carve((size_t)COUT * DCATP * 2);
    unsigned short* Wab     = (unsigned short*)carve((size_t)COUT * DAP * 2);
    unsigned short* act     = (unsigned short*)carve((size_t)PP * DAP * 2);     // ~112 MB
    float*          y1      = (float*)carve((size_t)COUT * PP * 4);             // 128 MB
    float*          y2      = (float*)carve((size_t)COUT * PP * 4);             // 128 MB

    // 1) centers -> ws + output
    wslfa_centers<<<(BMTOT + 255) / 256, 256, 0, stream>>>(xyz, centers, out);
    // 2) KNN (one wave per center)
    wslfa_knn<<<BMTOT, 32, 0, stream>>>(xyz, centers, idxknn);
    // 3) build bf16 activation rows (cat + zero padding)
    wslfa_cat<<<(PP + 255) / 256, 256, 0, stream>>>(xyz, feat, centers, idxknn, act);
    // 4) weights -> padded bf16
    wslfa_wprep<<<(COUT * DAP + 255) / 256, 256, 0, stream>>>(Wf, Wa, Wfb, Wab);
    // 5) GEMM1: y1 = W_feat x cat + b_feat   (chunks over 96 padded channels)
    wslfa_gemm<<<PP / 16, 256, 0, stream>>>(act, DAP, Wfb, DCATP, DCATP / 32, bf_, y1);
    // 6) BN stats on y1
    wslfa_bnstats<<<COUT, 256, 0, stream>>>(y1, mu1, is1);
    // 7) f_prime = relu(bn(y1)) in place
    wslfa_bnapply<<<(unsigned)(((size_t)COUT * PP) / 256), 256, 0, stream>>>(y1, mu1, is1, gf, bef);
    // 8) centered f_prime -> act[:, 67..194] (bf16)
    wslfa_center<<<(COUT * BMTOT) / 256, 256, 0, stream>>>(y1, act);
    // 9) GEMM2: y2 = W_alpha x alpha_in + b_alpha (chunks over 224 padded channels)
    wslfa_gemm<<<PP / 16, 256, 0, stream>>>(act, DAP, Wab, DAP, DAP / 32, ba_, y2);
    // 10) BN stats on y2
    wslfa_bnstats<<<COUT, 256, 0, stream>>>(y2, mu2, is2);
    // 11) fused alpha BN+ReLU + softmax(K) + weighted sum -> f_region
    wslfa_final<<<(COUT * BMTOT) / 8 / 32, 256, 0, stream>>>(y2, y1, mu2, is2, ga, bea, out);
}
